// ChessGNN_50912542326946
// MI455X (gfx1250) — matrix operand order repaired and checked by reading, not compiled, not run on hardware
//
#include <hip/hip_runtime.h>
#include <hip/hip_bf16.h>
#include <math.h>
#include <stdint.h>

// ---------------------------------------------------------------- constants
enum { Hh = 4, Cc = 64, Dd = 256, Bb = 2048, NPn = 65536, NSn = 131072,
       E1n = 524288, E2n = 1048576, POL = 5312 };

typedef __attribute__((ext_vector_type(16))) __bf16 v16bf;
typedef __attribute__((ext_vector_type(8)))  float  v8f;

// ---------------------------------------------------------------- embeddings
__global__ void embed_pieces_k(const float* __restrict__ px,
                               const float* __restrict__ w,
                               const float* __restrict__ b,
                               float* __restrict__ out) {
  int i = blockIdx.x * blockDim.x + threadIdx.x;
  if (i >= NPn * Dd) return;
  int n = i >> 8, d = i & 255;
  float acc = b[d];
#pragma unroll
  for (int k = 0; k < 12; ++k) acc += px[n * 12 + k] * w[k * Dd + d];
  out[i] = acc;
}

__global__ void embed_squares_k(const float* __restrict__ sx,
                                const float* __restrict__ w,
                                const float* __restrict__ b,
                                float* __restrict__ out) {
  int i = blockIdx.x * blockDim.x + threadIdx.x;
  if (i >= NSn * Dd) return;
  int n = i >> 8, d = i & 255;
  out[i] = sx[n] * w[d] + b[d];
}

// ------------------------------------------------- weight convert+transpose
// wt[n][k] = (bf16) w[k][n]; K is always 256 here.
__global__ void conv_wt_k(const float* __restrict__ w, __bf16* __restrict__ wt,
                          int N) {
  int i = blockIdx.x * blockDim.x + threadIdx.x;
  if (i >= N * Dd) return;
  int n = i >> 8, k = i & 255;
  wt[(size_t)n * Dd + k] = (__bf16)w[(size_t)k * N + n];
}

// ---------------------------------------------------------------- WMMA GEMM
// C[M,N] = A[M,K] @ W[K,N] (+bias)(+relu); W given pre-transposed bf16 [N][K].
// Block = 256 threads = 8 wave32 (4 in M x 2 in N); tile 64x64, K step 32.
// B tile staged via CDNA5 async global->LDS DMA (ASYNCcnt), A tile staged
// with in-register f32->bf16 conversion overlapping the DMA.
__global__ __launch_bounds__(256)
void gemm_wmma_bf16(const float* __restrict__ A, const __bf16* __restrict__ Wt,
                    const float* __restrict__ bias, float* __restrict__ C,
                    int M, int N, int K, int relu) {
  __shared__ __bf16 sA[64][32];   // [m][k]
  __shared__ __bf16 sB[64][32];   // [n][k]
  const int tid  = threadIdx.x;
  const int bm   = blockIdx.y * 64;
  const int bn   = blockIdx.x * 64;
  const int wave = tid >> 5;
  const int lane = tid & 31;
  const int hf   = lane >> 4;     // lane half
  const int l16  = lane & 15;
  const int wm   = wave >> 1;     // 0..3  (M sub-tile)
  const int wn   = wave & 1;      // 0..1  (N sub-tile, 32 wide)

  // this thread's B-tile DMA slot: one b128 (8 bf16) per K-step
  const int brow   = tid >> 2;        // 0..63  (n within tile)
  const int bchunk = (tid & 3) * 8;   // bf16 offset within the 32-long k row
  const unsigned ldsB = (unsigned)(uintptr_t)(&sB[brow][bchunk]);

  v8f acc0 = {0.f, 0.f, 0.f, 0.f, 0.f, 0.f, 0.f, 0.f};
  v8f acc1 = {0.f, 0.f, 0.f, 0.f, 0.f, 0.f, 0.f, 0.f};

  for (int k0 = 0; k0 < K; k0 += 32) {
    // --- issue async DMA of the bf16 weight tile into LDS (no VGPR transit)
    {
      const __bf16* gB = Wt + (size_t)(bn + brow) * K + (k0 + bchunk);
      asm volatile("global_load_async_to_lds_b128 %0, %1, off"
                   :: "v"(ldsB), "v"(gB) : "memory");
    }
    // --- stage A 64x32 as bf16 (8 elems / thread), overlaps with the DMA
#pragma unroll
    for (int i = 0; i < 8; ++i) {
      int idx = tid * 8 + i;
      int r = idx >> 5, c = idx & 31;
      sA[r][c] = (__bf16)A[(size_t)(bm + r) * K + (k0 + c)];
    }
    if (k0 + 32 < K) {  // CDNA5 global_prefetch of next A tile
      __builtin_prefetch(&A[(size_t)(bm + (tid >> 2)) * K + (k0 + 32)], 0, 1);
    }
    asm volatile("s_wait_asynccnt 0x0" ::: "memory");
    __syncthreads();

    // A fragment: 16x32 bf16 layout (ISA 7.12.2): V0..3 -> K 0..15 (+8*half),
    // V4..7 -> K 16..31 (+8*half); lanes 0..15 / 16..31 both carry M=l16.
    v16bf af;
    const int arow = wm * 16 + l16;
#pragma unroll
    for (int e = 0; e < 16; ++e) {
      int v = e >> 1, p = e & 1;
      int k = (v < 4) ? (v * 2 + p + hf * 8) : (16 + (v - 4) * 2 + p + hf * 8);
      af[e] = sA[arow][k];
    }
    // B fragments: 32x16 bf16, N = l16, elems 0..7 -> K 0..7 (+8*half), 8..15 -> K 16..23 (+8*half)
    v16bf bf0, bf1;
#pragma unroll
    for (int e = 0; e < 16; ++e) {
      int k = (e < 8) ? (hf * 8 + e) : (16 + hf * 8 + (e - 8));
      bf0[e] = sB[wn * 32 + l16][k];
      bf1[e] = sB[wn * 32 + 16 + l16][k];
    }
    acc0 = __builtin_amdgcn_wmma_f32_16x16x32_bf16(false, af, false, bf0,
                                                   (short)0, acc0, false, false);
    acc1 = __builtin_amdgcn_wmma_f32_16x16x32_bf16(false, af, false, bf1,
                                                   (short)0, acc1, false, false);
    __syncthreads();
  }

  // D layout: VGPR r -> M = r + 8*half ; N = l16
#pragma unroll
  for (int r = 0; r < 8; ++r) {
    int row  = bm + wm * 16 + hf * 8 + r;
    int col0 = bn + wn * 32 + l16;
    int col1 = col0 + 16;
    float v0 = acc0[r] + (bias ? bias[col0] : 0.f);
    float v1 = acc1[r] + (bias ? bias[col1] : 0.f);
    if (relu) { v0 = fmaxf(v0, 0.f); v1 = fmaxf(v1, 0.f); }
    C[(size_t)row * N + col0] = v0;
    C[(size_t)row * N + col1] = v1;
  }
}

// ---------------------------------------------------------------- GAT pieces
__global__ void node_alpha_k(const float* __restrict__ x,
                             const float* __restrict__ avec,
                             float* __restrict__ al, int Nnodes) {
  int i = blockIdx.x * blockDim.x + threadIdx.x;
  if (i >= Nnodes * Hh) return;
  int n = i >> 2, h = i & 3;
  const float* xr = x + (size_t)n * Dd + h * Cc;
  const float* ar = avec + h * Cc;
  float s = 0.f;
#pragma unroll 8
  for (int c = 0; c < Cc; ++c) s += xr[c] * ar[c];
  al[i] = s;
}

__global__ void init_seg_k(float* __restrict__ m, float* __restrict__ ss, int n) {
  int i = blockIdx.x * blockDim.x + threadIdx.x;
  if (i < n) { m[i] = -__builtin_inff(); ss[i] = 0.f; }
}

__global__ void init_out_bias_k(float* __restrict__ out,
                                const float* __restrict__ bias, int nNodes) {
  int i = blockIdx.x * blockDim.x + threadIdx.x;
  if (i < nNodes * Dd) out[i] = bias[i & 255];
}

__device__ inline void atomicMaxF(float* addr, float v) {
  if (v >= 0.f) atomicMax((int*)addr, __float_as_int(v));
  else          atomicMin((unsigned int*)addr, __float_as_uint(v));
}

__global__ void edge_max_k(const int* __restrict__ src, const int* __restrict__ dst,
                           const float* __restrict__ als, const float* __restrict__ ald,
                           float* __restrict__ m, int E) {
  int i = blockIdx.x * blockDim.x + threadIdx.x;
  if (i >= E * Hh) return;
  int e = i >> 2, h = i & 3;
  float v = als[src[e] * Hh + h] + ald[dst[e] * Hh + h];
  v = (v > 0.f) ? v : 0.2f * v;
  atomicMaxF(&m[dst[e] * Hh + h], v);
}

__global__ void edge_exp_k(const int* __restrict__ src, const int* __restrict__ dst,
                           const float* __restrict__ als, const float* __restrict__ ald,
                           const float* __restrict__ m, float* __restrict__ eexp,
                           float* __restrict__ ssum, int E) {
  int i = blockIdx.x * blockDim.x + threadIdx.x;
  if (i >= E * Hh) return;
  int e = i >> 2, h = i & 3;
  int dh = dst[e] * Hh + h;
  float v = als[src[e] * Hh + h] + ald[dh];
  v = (v > 0.f) ? v : 0.2f * v;
  float w = __expf(v - m[dh]);
  eexp[i] = w;
  atomicAdd(&ssum[dh], w);
}

// one thread per (edge, head, 16-channel chunk)
__global__ void edge_aggr_k(const int* __restrict__ src, const int* __restrict__ dst,
                            const float* __restrict__ eexp, const float* __restrict__ ssum,
                            const float* __restrict__ xsrc, float* __restrict__ out, int E) {
  int i = blockIdx.x * blockDim.x + threadIdx.x;
  if (i >= E * Hh * 4) return;
  int e   = i >> 4;
  int h   = (i >> 2) & 3;
  int cb  = (i & 3) * 16;
  int d   = dst[e], s = src[e];
  float alpha = eexp[e * Hh + h] / (ssum[d * Hh + h] + 1e-16f);
  const float* xr = xsrc + (size_t)s * Dd + h * Cc + cb;
  float*       o  = out  + (size_t)d * Dd + h * Cc + cb;
#pragma unroll
  for (int c = 0; c < 16; ++c) atomicAdd(&o[c], alpha * xr[c]);
}

// wave32-per-node LayerNorm + ReLU
__global__ __launch_bounds__(256)
void ln_relu_k(const float* __restrict__ in, const float* __restrict__ g,
               const float* __restrict__ b, float* __restrict__ out, int Nnodes) {
  int wave = threadIdx.x >> 5, lane = threadIdx.x & 31;
  int n = blockIdx.x * 8 + wave;
  if (n >= Nnodes) return;
  const float* x = in + (size_t)n * Dd;
  float v[8]; float s = 0.f;
#pragma unroll
  for (int i = 0; i < 8; ++i) { v[i] = x[lane + i * 32]; s += v[i]; }
#pragma unroll
  for (int off = 16; off > 0; off >>= 1) s += __shfl_xor(s, off, 32);
  float mu = s * (1.f / 256.f);
  float var = 0.f;
#pragma unroll
  for (int i = 0; i < 8; ++i) { float d = v[i] - mu; var += d * d; }
#pragma unroll
  for (int off = 16; off > 0; off >>= 1) var += __shfl_xor(var, off, 32);
  float rs = rsqrtf(var * (1.f / 256.f) + 1e-5f);
#pragma unroll
  for (int i = 0; i < 8; ++i) {
    int d = lane + i * 32;
    float y = (v[i] - mu) * rs * g[d] + b[d];
    out[(size_t)n * Dd + d] = fmaxf(y, 0.f);
  }
}

// sq_batch is repeat(arange(B), 64): graph b owns nodes [64b, 64b+64)
__global__ void pool_mean_k(const float* __restrict__ s, float* __restrict__ g) {
  int i = blockIdx.x * blockDim.x + threadIdx.x;
  if (i >= Bb * Dd) return;
  int b = i >> 8, d = i & 255;
  float acc = 0.f;
#pragma unroll 8
  for (int j = 0; j < 64; ++j) acc += s[((size_t)b * 64 + j) * Dd + d];
  g[i] = acc * (1.f / 64.f);
}

// value = tanh(h @ w_v2 + b_v2), wave32 per graph
__global__ __launch_bounds__(256)
void value_out_k(const float* __restrict__ h, const float* __restrict__ w,
                 const float* __restrict__ b2, float* __restrict__ out) {
  int wave = threadIdx.x >> 5, lane = threadIdx.x & 31;
  int b = blockIdx.x * 8 + wave;
  if (b >= Bb) return;
  float s = 0.f;
#pragma unroll
  for (int i = 0; i < 8; ++i) { int d = lane + i * 32; s += h[(size_t)b * Dd + d] * w[d]; }
#pragma unroll
  for (int off = 16; off > 0; off >>= 1) s += __shfl_xor(s, off, 32);
  if (lane == 0) out[b] = tanhf(s + b2[0]);
}

// ---------------------------------------------------------------- launcher
static inline dim3 g1(int n, int blk = 256) { return dim3((n + blk - 1) / blk); }

extern "C" void kernel_launch(void* const* d_in, const int* in_sizes, int n_in,
                              void* d_out, int out_size, void* d_ws, size_t ws_size,
                              hipStream_t stream) {
  (void)in_sizes; (void)n_in; (void)out_size; (void)ws_size;
  const float* piece_x = (const float*)d_in[0];
  const float* square_x = (const float*)d_in[1];
  const int*   ps_src = (const int*)d_in[2];
  const int*   ps_dst = (const int*)d_in[3];
  const int*   ss_src = (const int*)d_in[4];
  const int*   ss_dst = (const int*)d_in[5];
  // d_in[6] = sq_batch (contiguity exploited directly)
  const float* w_pe = (const float*)d_in[7];   const float* b_pe = (const float*)d_in[8];
  const float* w_se = (const float*)d_in[9];   const float* b_se = (const float*)d_in[10];
  const float* l0_wsrc = (const float*)d_in[11];
  const float* l0_wdst = (const float*)d_in[12];
  const float* l0_asrc = (const float*)d_in[13];
  const float* l0_adst = (const float*)d_in[14];
  const float* l0_bias = (const float*)d_in[15];
  const float* w_lin   = (const float*)d_in[16];   // [3,256,256]
  const float* a_src   = (const float*)d_in[17];   // [3,256]
  const float* a_dst   = (const float*)d_in[18];
  const float* gat_bias= (const float*)d_in[19];
  const float* ln_g    = (const float*)d_in[20];   // [4,256]
  const float* ln_b    = (const float*)d_in[21];
  const float* w_p1 = (const float*)d_in[22]; const float* b_p1 = (const float*)d_in[23];
  const float* w_p2 = (const float*)d_in[24]; const float* b_p2 = (const float*)d_in[25];
  const float* w_v1 = (const float*)d_in[26]; const float* b_v1 = (const float*)d_in[27];
  const float* w_v2 = (const float*)d_in[28]; const float* b_v2 = (const float*)d_in[29];

  float* policy = (float*)d_out;                    // [B,5312]
  float* value  = (float*)d_out + (size_t)Bb * POL; // [B,1]

  // workspace carving (fp32 region, then bf16 weight region)
  float* ws = (float*)d_ws;  size_t off = 0;
  auto carve = [&](size_t n) { float* p = ws + off; off += n; return p; };
  float* p    = carve((size_t)NPn * Dd);
  float* xs   = carve((size_t)NPn * Dd);
  float* scur = carve((size_t)NSn * Dd);
  float* xpj  = carve((size_t)NSn * Dd);
  float* sagg = carve((size_t)NSn * Dd);
  float* als  = carve((size_t)NSn * Hh);
  float* ald  = carve((size_t)NSn * Hh);
  float* mseg = carve((size_t)NSn * Hh);
  float* ssum = carve((size_t)NSn * Hh);
  float* eexp = carve((size_t)E2n * Hh);
  float* gvec = carve((size_t)Bb * Dd);
  float* h1   = carve((size_t)Bb * Dd);

  __bf16* wb = (__bf16*)(ws + off);  size_t boff = 0;
  auto carveb = [&](size_t n) { __bf16* q = wb + boff; boff += n; return q; };
  __bf16* t_l0s = carveb((size_t)Dd * Dd);
  __bf16* t_l0d = carveb((size_t)Dd * Dd);
  __bf16* t_lin[3]; for (int i = 0; i < 3; ++i) t_lin[i] = carveb((size_t)Dd * Dd);
  __bf16* t_p1  = carveb((size_t)Dd * Dd);
  __bf16* t_p2  = carveb((size_t)Dd * POL);
  __bf16* t_v1  = carveb((size_t)Dd * Dd);

  // weight convert+transpose (tiny: ~3.5 MB total, once per launch)
  conv_wt_k<<<g1(Dd * Dd), 256, 0, stream>>>(l0_wsrc, t_l0s, Dd);
  conv_wt_k<<<g1(Dd * Dd), 256, 0, stream>>>(l0_wdst, t_l0d, Dd);
  for (int i = 0; i < 3; ++i)
    conv_wt_k<<<g1(Dd * Dd), 256, 0, stream>>>(w_lin + (size_t)i * Dd * Dd, t_lin[i], Dd);
  conv_wt_k<<<g1(Dd * Dd), 256, 0, stream>>>(w_p1, t_p1, Dd);
  conv_wt_k<<<g1(Dd * POL), 256, 0, stream>>>(w_p2, t_p2, POL);
  conv_wt_k<<<g1(Dd * Dd), 256, 0, stream>>>(w_v1, t_v1, Dd);

  // embeddings
  embed_pieces_k<<<g1(NPn * Dd), 256, 0, stream>>>(piece_x, w_pe, b_pe, p);
  embed_squares_k<<<g1(NSn * Dd), 256, 0, stream>>>(square_x, w_se, b_se, scur);

  // ---- layer 0: bipartite piece -> square
  gemm_wmma_bf16<<<dim3(Dd / 64, NPn / 64), 256, 0, stream>>>(p, t_l0s, nullptr, xs,
                                                              NPn, Dd, Dd, 0);
  gemm_wmma_bf16<<<dim3(Dd / 64, NSn / 64), 256, 0, stream>>>(scur, t_l0d, nullptr, xpj,
                                                              NSn, Dd, Dd, 0);
  node_alpha_k<<<g1(NPn * Hh), 256, 0, stream>>>(xs, l0_asrc, als, NPn);
  node_alpha_k<<<g1(NSn * Hh), 256, 0, stream>>>(xpj, l0_adst, ald, NSn);
  init_seg_k<<<g1(NSn * Hh), 256, 0, stream>>>(mseg, ssum, NSn * Hh);
  init_out_bias_k<<<g1(NSn * Dd), 256, 0, stream>>>(sagg, l0_bias, NSn);
  edge_max_k<<<g1(E1n * Hh), 256, 0, stream>>>(ps_src, ps_dst, als, ald, mseg, E1n);
  edge_exp_k<<<g1(E1n * Hh), 256, 0, stream>>>(ps_src, ps_dst, als, ald, mseg, eexp, ssum, E1n);
  edge_aggr_k<<<g1(E1n * Hh * 4), 256, 0, stream>>>(ps_src, ps_dst, eexp, ssum, xs, sagg, E1n);
  ln_relu_k<<<g1(NSn * 32), 256, 0, stream>>>(sagg, ln_g, ln_b, scur, NSn);

  // ---- layers 1..3: square -> square
  for (int i = 0; i < 3; ++i) {
    gemm_wmma_bf16<<<dim3(Dd / 64, NSn / 64), 256, 0, stream>>>(scur, t_lin[i], nullptr,
                                                                xpj, NSn, Dd, Dd, 0);
    node_alpha_k<<<g1(NSn * Hh), 256, 0, stream>>>(xpj, a_src + i * Dd, als, NSn);
    node_alpha_k<<<g1(NSn * Hh), 256, 0, stream>>>(xpj, a_dst + i * Dd, ald, NSn);
    init_seg_k<<<g1(NSn * Hh), 256, 0, stream>>>(mseg, ssum, NSn * Hh);
    init_out_bias_k<<<g1(NSn * Dd), 256, 0, stream>>>(sagg, gat_bias + i * Dd, NSn);
    edge_max_k<<<g1(E2n * Hh), 256, 0, stream>>>(ss_src, ss_dst, als, ald, mseg, E2n);
    edge_exp_k<<<g1(E2n * Hh), 256, 0, stream>>>(ss_src, ss_dst, als, ald, mseg, eexp, ssum, E2n);
    edge_aggr_k<<<g1(E2n * Hh * 4), 256, 0, stream>>>(ss_src, ss_dst, eexp, ssum, xpj, sagg, E2n);
    ln_relu_k<<<g1(NSn * 32), 256, 0, stream>>>(sagg, ln_g + (i + 1) * Dd, ln_b + (i + 1) * Dd,
                                                scur, NSn);
  }

  // ---- pooling + heads
  pool_mean_k<<<g1(Bb * Dd), 256, 0, stream>>>(scur, gvec);
  gemm_wmma_bf16<<<dim3(Dd / 64, Bb / 64), 256, 0, stream>>>(gvec, t_p1, b_p1, h1,
                                                             Bb, Dd, Dd, 1);
  gemm_wmma_bf16<<<dim3(POL / 64, Bb / 64), 256, 0, stream>>>(h1, t_p2, b_p2, policy,
                                                              Bb, POL, Dd, 0);
  gemm_wmma_bf16<<<dim3(Dd / 64, Bb / 64), 256, 0, stream>>>(gvec, t_v1, b_v1, h1,
                                                             Bb, Dd, Dd, 1);
  value_out_k<<<g1(Bb * 32), 256, 0, stream>>>(h1, w_v2, b_v2, value);
}
